// WindowAttention_3298534884159
// MI455X (gfx1250) — matrix operand order repaired and checked by reading, not compiled, not run on hardware
//
#include <hip/hip_runtime.h>
#include <hip/hip_bf16.h>
#include <math.h>

typedef _Float16 v8h  __attribute__((ext_vector_type(8)));
typedef _Float16 v16h __attribute__((ext_vector_type(16)));
typedef float    v8f  __attribute__((ext_vector_type(8)));

// ---------------------------------------------------------------------------
// Problem constants
// ---------------------------------------------------------------------------
#define BATCH   64
#define SEQ     2048
#define CDIM    256
#define HEADS   8
#define CHEAD   32
#define MTOK    (BATCH * SEQ)          // 131072 rows
#define QKVN    (3 * CDIM)             // 768

// ---------------------------------------------------------------------------
// f32 -> f16 conversion
// ---------------------------------------------------------------------------
__global__ void convert_f32_to_f16(const float* __restrict__ in,
                                   _Float16* __restrict__ out, int n) {
    int i = blockIdx.x * blockDim.x + threadIdx.x;
    if (i < n) out[i] = (_Float16)in[i];
}

// ---------------------------------------------------------------------------
// WMMA fragment loaders (wave32, 16-bit operands, contiguous k order in both A and B)
// A 16x32 (MxK): lane l -> row l&15, K = (l>>4)*16 .. +15 contiguous
// B 32x16 (KxN): lane l -> col l&15, K = (l>>4)*16 .. +15 contiguous
// ---------------------------------------------------------------------------
__device__ __forceinline__ v16h load_a_frag(const _Float16* __restrict__ A,
                                            int lda, int row0, int k0, int lane) {
    const int r  = row0 + (lane & 15);
    const int hs = lane >> 4;
    const _Float16* p = A + (size_t)r * lda + k0 + hs * 16;
    v8h c0 = *(const v8h*)(p);
    v8h c1 = *(const v8h*)(p + 8);
    v16h a;
#pragma unroll
    for (int i = 0; i < 8; ++i) { a[i] = c0[i]; a[i + 8] = c1[i]; }
    return a;
}

__device__ __forceinline__ v16h load_b_frag(const _Float16* __restrict__ W,
                                            int ldb, int col0, int k0, int lane) {
    const int n  = col0 + (lane & 15);
    const int kb = (lane >> 4) * 16;
    return *(const v16h*)(W + (size_t)n * ldb + k0 + kb);
}

// ---------------------------------------------------------------------------
// GEMM: D[M x N] = A[M x K] * W^T (W is [N x K] row-major) + bias
// One wave computes a 16x64 tile. K fully unrolled (K = 256 -> 8 WMMA steps).
// ---------------------------------------------------------------------------
template <typename OutT>
__global__ void gemm_wmma_f16(const _Float16* __restrict__ A,
                              const _Float16* __restrict__ W,
                              const float* __restrict__ bias,
                              OutT* __restrict__ D,
                              int M, int N, int K) {
    const int lane = threadIdx.x & 31;
    const int wid  = (blockIdx.x * blockDim.x + threadIdx.x) >> 5;

    const int ntiles4 = N >> 6;                 // number of 64-wide column groups
    const int mt  = wid / ntiles4;              // 16-row tile index
    const int nt4 = wid % ntiles4;              // 64-col group index
    if (mt * 16 >= M) return;

    const int row0 = mt * 16;
    const int col0 = nt4 * 64;

    v8f acc0 = {}, acc1 = {}, acc2 = {}, acc3 = {};

#pragma unroll
    for (int k0 = 0; k0 < CDIM; k0 += 32) {
        v16h a  = load_a_frag(A, K, row0, k0, lane);
        v16h b0 = load_b_frag(W, K, col0 +  0, k0, lane);
        v16h b1 = load_b_frag(W, K, col0 + 16, k0, lane);
        v16h b2 = load_b_frag(W, K, col0 + 32, k0, lane);
        v16h b3 = load_b_frag(W, K, col0 + 48, k0, lane);
        acc0 = __builtin_amdgcn_wmma_f32_16x16x32_f16(false, a, false, b0, (short)0, acc0, false, false);
        acc1 = __builtin_amdgcn_wmma_f32_16x16x32_f16(false, a, false, b1, (short)0, acc1, false, false);
        acc2 = __builtin_amdgcn_wmma_f32_16x16x32_f16(false, a, false, b2, (short)0, acc2, false, false);
        acc3 = __builtin_amdgcn_wmma_f32_16x16x32_f16(false, a, false, b3, (short)0, acc3, false, false);
    }

    // Epilogue: C/D layout -> lane l, vgpr v : row = v + 8*(l>>4), col = l&15
    const int hs = lane >> 4;
    const int cn = lane & 15;
#pragma unroll
    for (int j = 0; j < 4; ++j) {
        const v8f* accp = (j == 0) ? &acc0 : (j == 1) ? &acc1 : (j == 2) ? &acc2 : &acc3;
        const int coln = col0 + j * 16 + cn;
        const float bv = bias[coln];
#pragma unroll
        for (int v = 0; v < 8; ++v) {
            const int row = row0 + v + 8 * hs;
            D[(size_t)row * N + coln] = (OutT)((*accp)[v] + bv);
        }
    }
}

// ---------------------------------------------------------------------------
// Per-token head attention. One wave32 per token, lane = channel (Ch == 32).
// qkv row layout: [3][H][Ch] f16.  Output written as f16 in the permuted
// (B, H, N, Ch) layout == row-major (M x C) input for the projection GEMM.
// ---------------------------------------------------------------------------
__device__ __forceinline__ float wave_reduce_add(float v) {
#pragma unroll
    for (int off = 16; off > 0; off >>= 1) v += __shfl_xor(v, off, 32);
    return v;
}

__global__ void head_attention(const _Float16* __restrict__ qkv,
                               _Float16* __restrict__ attn_out) {
    const int lane = threadIdx.x & 31;                       // channel
    const int tok  = (blockIdx.x * blockDim.x + threadIdx.x) >> 5;
    const int b = tok >> 11;                                 // / SEQ
    const int n = tok & (SEQ - 1);

    const _Float16* row = qkv + (size_t)tok * QKVN;
    float qf[HEADS], kf[HEADS], vf[HEADS];
#pragma unroll
    for (int h = 0; h < HEADS; ++h) {
        qf[h] = (float)row[          h * CHEAD + lane];
        kf[h] = (float)row[CDIM    + h * CHEAD + lane];
        vf[h] = (float)row[2*CDIM  + h * CHEAD + lane];
    }

    const float scale = 0.17677669529663687f;                // 32^-0.5

#pragma unroll
    for (int h = 0; h < HEADS; ++h) {
        float s[HEADS];
#pragma unroll
        for (int g = 0; g < HEADS; ++g) s[g] = qf[h] * kf[g];
        // butterfly-reduce all 8 scores across the 32 channels
#pragma unroll
        for (int off = 16; off > 0; off >>= 1) {
#pragma unroll
            for (int g = 0; g < HEADS; ++g) s[g] += __shfl_xor(s[g], off, 32);
        }
        float m = s[0] * scale;
#pragma unroll
        for (int g = 0; g < HEADS; ++g) { s[g] *= scale; m = fmaxf(m, s[g]); }
        float e[HEADS], sum = 0.0f;
#pragma unroll
        for (int g = 0; g < HEADS; ++g) { e[g] = __expf(s[g] - m); sum += e[g]; }
        const float inv = 1.0f / sum;
        float o = 0.0f;
#pragma unroll
        for (int g = 0; g < HEADS; ++g) o += e[g] * vf[g];
        o *= inv;
        // permuted layout: idx = b*(H*N*Ch) + h*(N*Ch) + n*Ch + c
        attn_out[(size_t)b * (HEADS * SEQ * CHEAD) + (size_t)h * (SEQ * CHEAD)
                 + (size_t)n * CHEAD + lane] = (_Float16)o;
    }
}

// ---------------------------------------------------------------------------
// Launch
// ---------------------------------------------------------------------------
extern "C" void kernel_launch(void* const* d_in, const int* in_sizes, int n_in,
                              void* d_out, int out_size, void* d_ws, size_t ws_size,
                              hipStream_t stream) {
    const float* x      = (const float*)d_in[0];   // (64,2048,256)
    const float* qkv_w  = (const float*)d_in[1];   // (768,256)
    const float* qkv_b  = (const float*)d_in[2];   // (768,)
    const float* proj_w = (const float*)d_in[3];   // (256,256)
    const float* proj_b = (const float*)d_in[4];   // (256,)
    float* out = (float*)d_out;                    // (64,2048,256) f32

    // workspace carve (all 256-B aligned by construction)
    char* p = (char*)d_ws;
    _Float16* xh     = (_Float16*)p; p += (size_t)MTOK * CDIM * 2;      //  67 MB
    _Float16* qkvwh  = (_Float16*)p; p += (size_t)QKVN * CDIM * 2;      // 384 KB
    _Float16* projwh = (_Float16*)p; p += (size_t)CDIM * CDIM * 2;      // 128 KB
    _Float16* qkvh   = (_Float16*)p; p += (size_t)MTOK * QKVN * 2;      // 201 MB
    _Float16* attnh  = (_Float16*)p; p += (size_t)MTOK * CDIM * 2;      //  67 MB

    // 1) precision conversion
    {
        int n = MTOK * CDIM;
        convert_f32_to_f16<<<(n + 255) / 256, 256, 0, stream>>>(x, xh, n);
        n = QKVN * CDIM;
        convert_f32_to_f16<<<(n + 255) / 256, 256, 0, stream>>>(qkv_w, qkvwh, n);
        n = CDIM * CDIM;
        convert_f32_to_f16<<<(n + 255) / 256, 256, 0, stream>>>(proj_w, projwh, n);
    }

    // 2) QKV GEMM: (131072 x 256) @ (256 x 768) + b -> f16
    {
        const int waves  = (MTOK / 16) * (QKVN / 64);   // 98304
        const int blocks = waves / 8;                   // 12288 (256-thread blocks)
        gemm_wmma_f16<_Float16><<<blocks, 256, 0, stream>>>(
            xh, qkvwh, qkv_b, qkvh, MTOK, QKVN, CDIM);
    }

    // 3) per-token 8x8 head attention, writes permuted f16
    {
        const int blocks = MTOK / 8;                    // one wave per token
        head_attention<<<blocks, 256, 0, stream>>>(qkvh, attnh);
    }

    // 4) projection GEMM: (131072 x 256) @ (256 x 256) + b -> f32 output
    {
        const int waves  = (MTOK / 16) * (CDIM / 64);   // 32768
        const int blocks = waves / 8;                   // 4096
        gemm_wmma_f16<float><<<blocks, 256, 0, stream>>>(
            attnh, projwh, proj_b, out, MTOK, CDIM, CDIM);
    }
}
